// IRLLAYERS_45827301048479
// MI455X (gfx1250) — compile-verified, matrix operand order checked by reading, not run on hardware
//
#include <hip/hip_runtime.h>
#include <hip/hip_bf16.h>

#define BB 16
#define SS 2048
#define DD 128
#define UU 128
#define ROWP 136  // padded LDS row length in bf16 (272 B = 68 dwords -> conflict-free)

typedef __bf16 bf16_t;
typedef __attribute__((ext_vector_type(16))) __bf16 v16bf;
typedef __attribute__((ext_vector_type(8)))  float  v8f;
typedef __attribute__((ext_vector_type(4))) unsigned int v4u;
typedef __attribute__((ext_vector_type(8))) int v8i;
typedef __attribute__((ext_vector_type(4))) int v4i;

union FragU { v16bf v; uint4 q[2]; };

#ifndef __has_builtin
#define __has_builtin(x) 0
#endif
#if __has_builtin(__builtin_amdgcn_tensor_load_to_lds) && \
    __has_builtin(__builtin_amdgcn_s_wait_tensorcnt)
#define HAVE_TDM 1
#else
#define HAVE_TDM 0
#endif

// Load one 16x32 bf16 WMMA operand fragment; rowbase points at element [row][0]
// of a row-major matrix (any row stride; only within-row offsets used here).
// Lane layout per ISA 7.12.2: lane L (m=L%16, kh=L/16) holds
// K = kh*8 + {0..7} and 16 + kh*8 + {0..7}  (plus k0 base).
__device__ __forceinline__ v16bf load_frag(const bf16_t* rowbase, int k0, int kh) {
  const bf16_t* p = rowbase + k0 + kh * 8;
  FragU f;
  f.q[0] = *(const uint4*)(p);
  f.q[1] = *(const uint4*)(p + 16);
  return f.v;
}

#if HAVE_TDM
// TDM: async-copy one 128x128 bf16 tile (row stride 128 elements in global)
// into LDS, inserting 16 B of padding after every 256 B row so LDS rows have
// stride ROWP (272 B). Issued by one wave; tracked with TENSORcnt.
__device__ __forceinline__ void tdm_load_tile128(const void* gsrc, void* ldst) {
  unsigned long long ga = (unsigned long long)(size_t)gsrc;
  unsigned int lds_off = (unsigned int)(size_t)ldst;  // low 32 bits = LDS offset
  // D# group0: count=1 | lds_addr | global_addr[56:0] | type=2
  v4u g0 = { 1u, lds_off, (unsigned int)ga,
             (unsigned int)((ga >> 32) & 0x1FFFFFFull) | 0x80000000u };
  // D# group1:
  //  w0: data_size=1(2B)<<16 | pad_enable<<20 | pad_interval=5(256B)<<22 | pad_amount=3(16B)<<25
  //  w1: tensor_dim0 lo16 (128) << 16
  //  w2: tensor_dim0 hi16 (0) | tensor_dim1 lo16 (2048) << 16
  //  w3: tensor_dim1 hi16 (0) | tile_dim0 (128) << 16
  //  w4: tile_dim1 (128) | tile_dim2 (0) << 16
  //  w5: tensor_dim0_stride lo32 (128) ; w6,w7: hi bits / dim1_stride = 0
  v8i g1 = { (int)0x07510000, (int)0x00800000, (int)0x08000000, (int)0x00800000,
             128, 128, 0, 0 };
  v4i gz = { 0, 0, 0, 0 };
#if __clang_major__ >= 23
  v8i gz8 = { 0, 0, 0, 0, 0, 0, 0, 0 };
  __builtin_amdgcn_tensor_load_to_lds(g0, g1, gz, gz, gz8, 0);
#else
  __builtin_amdgcn_tensor_load_to_lds(g0, g1, gz, gz, 0);
#endif
}
#define STAGE_TILE(gptr, dstbuf)                                   \
  do { if (tid < 32) tdm_load_tile128((gptr), &(dstbuf)[0][0]); } while (0)
#define STAGE_WAIT(more)                                           \
  do { if (tid < 32) {                                             \
         if (more) __builtin_amdgcn_s_wait_tensorcnt(1);           \
         else      __builtin_amdgcn_s_wait_tensorcnt(0);           \
       } } while (0)
#else
#define STAGE_TILE(gptr, dstbuf)                                   \
  do { for (int _i = tid; _i < 128 * 16; _i += 256) {              \
         int _r = _i >> 4, _c = (_i & 15) * 8;                     \
         *(uint4*)&(dstbuf)[_r][_c] =                              \
             *(const uint4*)((gptr) + _r * UU + _c);               \
       } } while (0)
#define STAGE_WAIT(more) do { } while (0)
#endif

// ---------------- K0: time factor = sigmoid(theta - mu * t) ----------------
__global__ __launch_bounds__(256) void timefact_kernel(
    const float* __restrict__ t, const float* __restrict__ theta,
    const float* __restrict__ mu, float* __restrict__ tf) {
  int idx = blockIdx.x * 256 + threadIdx.x;
  if (idx < BB * SS) {
    int s = idx & (SS - 1);
    float z = theta[s] - mu[s] * t[idx];
    tf[idx] = 1.0f / (1.0f + __expf(-z));
  }
}

// ---------------- K1: q = x@Wq, k = x@Wk (bf16 out) ----------------
// grid: (S/128, B, 2); block 256 (8 waves). 128x128 output tile / WG.
__global__ __launch_bounds__(256) void qk_proj_kernel(
    const float* __restrict__ x, const float* __restrict__ Wq,
    const float* __restrict__ Wk, bf16_t* __restrict__ qk_ws) {
  __shared__ __align__(16) bf16_t xs[128][128];   // x tile, row-major (row, d)
  __shared__ __align__(16) bf16_t wsT[128][128];  // W transposed: (u, d)
  const int rb = blockIdx.x, b = blockIdx.y, z = blockIdx.z;
  const int tid = threadIdx.x;
  const float* xb = x + ((size_t)b * SS + (size_t)rb * 128) * DD;
  const float* Wp = (z == 0) ? Wq : Wk;
  for (int idx = tid; idx < 128 * 128 / 4; idx += 256) {
    int r = (idx * 4) >> 7, c = (idx * 4) & 127;
    float4 v = *(const float4*)(xb + r * DD + c);
    xs[r][c + 0] = (bf16_t)v.x; xs[r][c + 1] = (bf16_t)v.y;
    xs[r][c + 2] = (bf16_t)v.z; xs[r][c + 3] = (bf16_t)v.w;
    float4 w = *(const float4*)(Wp + r * UU + c);
    wsT[c + 0][r] = (bf16_t)w.x; wsT[c + 1][r] = (bf16_t)w.y;
    wsT[c + 2][r] = (bf16_t)w.z; wsT[c + 3][r] = (bf16_t)w.w;
  }
  __syncthreads();
  const int wave = tid >> 5, lane = tid & 31;
  const int lm = lane & 15, kh = lane >> 4;
  const int m0 = wave * 16;
  v16bf aF[4];
#pragma unroll
  for (int ks = 0; ks < 4; ++ks)
    aF[ks] = load_frag(&xs[m0 + lm][0], ks * 32, kh);
  bf16_t* outp = qk_ws + ((size_t)z * BB + b) * SS * UU + (size_t)rb * 128 * UU;
#pragma unroll
  for (int n0 = 0; n0 < 8; ++n0) {
    v8f acc = {};
#pragma unroll
    for (int ks = 0; ks < 4; ++ks) {
      v16bf bF = load_frag(&wsT[n0 * 16 + lm][0], ks * 32, kh);
      acc = __builtin_amdgcn_wmma_f32_16x16x32_bf16(
          false, aF[ks], false, bF, (short)0, acc, false, false);
    }
#pragma unroll
    for (int r = 0; r < 8; ++r) {
      int srow = m0 + r + 8 * kh;  // C layout: VGPR r, half kh -> row r+8*kh
      outp[(size_t)srow * UU + n0 * 16 + lm] = (bf16_t)acc[r];
    }
  }
}

// ---------------- K2: per-row softmax stats (m_i, coef_i = tf_i / Z_i) -----
// grid: (S/128, B); block 256. Wave owns 16 rows; k tiles TDM-staged in LDS
// (double buffered) and shared by all 8 waves.
__global__ __launch_bounds__(256) void row_softmax_stats_kernel(
    const bf16_t* __restrict__ qk_ws, const float* __restrict__ tf,
    float* __restrict__ mrow, float* __restrict__ coef) {
  __shared__ __align__(16) bf16_t kbuf[2][128][ROWP];  // 2 x 34 KB
  const int rb = blockIdx.x, b = blockIdx.y;
  const int tid = threadIdx.x, wave = tid >> 5, lane = tid & 31;
  const int lm = lane & 15, kh = lane >> 4;
  const int i0 = rb * 128 + wave * 16;
  const bf16_t* qb = qk_ws + (size_t)b * SS * UU;
  const bf16_t* kb = qk_ws + ((size_t)BB + b) * SS * UU;
  const float scale = 0.08838834764831845f;  // 1/sqrt(128)
  v16bf aF[4];
#pragma unroll
  for (int ks = 0; ks < 4; ++ks)
    aF[ks] = load_frag(qb + (size_t)(i0 + lm) * UU, ks * 32, kh);
  float mr[8], zr[8];
#pragma unroll
  for (int r = 0; r < 8; ++r) { mr[r] = -1e30f; zr[r] = 0.0f; }

  STAGE_TILE(kb, kbuf[0]);
  for (int jb = 0; jb < SS / 128; ++jb) {
    const int p = jb & 1;
    if (jb + 1 < SS / 128)
      STAGE_TILE(kb + (size_t)(jb + 1) * 128 * UU, kbuf[p ^ 1]);
    STAGE_WAIT(jb + 1 < SS / 128);
    __syncthreads();
#pragma unroll
    for (int jt = 0; jt < 8; ++jt) {
      v8f acc = {};
#pragma unroll
      for (int ks = 0; ks < 4; ++ks) {
        // scores B operand: B[u][j] = k[j][u] -> rows of k, contiguous u
        v16bf bF = load_frag(&kbuf[p][jt * 16 + lm][0], ks * 32, kh);
        acc = __builtin_amdgcn_wmma_f32_16x16x32_bf16(
            false, aF[ks], false, bF, (short)0, acc, false, false);
      }
#pragma unroll
      for (int r = 0; r < 8; ++r) {
        float sv = acc[r] * scale;
        float tmax = sv;  // reduce over the 16 lanes holding this row
        tmax = fmaxf(tmax, __shfl_xor(tmax, 1, 32));
        tmax = fmaxf(tmax, __shfl_xor(tmax, 2, 32));
        tmax = fmaxf(tmax, __shfl_xor(tmax, 4, 32));
        tmax = fmaxf(tmax, __shfl_xor(tmax, 8, 32));
        float nm = fmaxf(mr[r], tmax);
        float pv = __expf(sv - nm);
        float ps = pv;
        ps += __shfl_xor(ps, 1, 32);
        ps += __shfl_xor(ps, 2, 32);
        ps += __shfl_xor(ps, 4, 32);
        ps += __shfl_xor(ps, 8, 32);
        zr[r] = zr[r] * __expf(mr[r] - nm) + ps;
        mr[r] = nm;
      }
    }
    __syncthreads();
  }
  if (lm == 0) {  // lanes 0 and 16 write their half's 8 rows
#pragma unroll
    for (int r = 0; r < 8; ++r) {
      int row = i0 + kh * 8 + r;
      mrow[b * SS + row] = mr[r];
      coef[b * SS + row] = tf[b * SS + row] / zr[r];
    }
  }
}

// ---------------- K3: w_j = sum_i coef_i * exp(s_ij - m_i) ----------------
// grid: (S/128, B); block 256. Wave owns 16 columns (B frags loop-invariant
// in registers); q tiles TDM-staged in LDS, shared by all 8 waves.
__global__ __launch_bounds__(256) void col_weight_kernel(
    const bf16_t* __restrict__ qk_ws, const float* __restrict__ mrow,
    const float* __restrict__ coef, float* __restrict__ wcol) {
  __shared__ __align__(16) bf16_t qbuf[2][128][ROWP];
  const int cb = blockIdx.x, b = blockIdx.y;
  const int tid = threadIdx.x, wave = tid >> 5, lane = tid & 31;
  const int lm = lane & 15, kh = lane >> 4;
  const int j0 = cb * 128 + wave * 16;
  const bf16_t* qb = qk_ws + (size_t)b * SS * UU;
  const bf16_t* kb = qk_ws + ((size_t)BB + b) * SS * UU;
  const float scale = 0.08838834764831845f;
  v16bf bF[4];  // loop-invariant B fragments (k rows for our columns)
#pragma unroll
  for (int ks = 0; ks < 4; ++ks)
    bF[ks] = load_frag(kb + (size_t)(j0 + lm) * UU, ks * 32, kh);
  float wloc = 0.0f;

  STAGE_TILE(qb, qbuf[0]);
  for (int ib = 0; ib < SS / 128; ++ib) {
    const int p = ib & 1;
    if (ib + 1 < SS / 128)
      STAGE_TILE(qb + (size_t)(ib + 1) * 128 * UU, qbuf[p ^ 1]);
    STAGE_WAIT(ib + 1 < SS / 128);
    __syncthreads();
#pragma unroll
    for (int it = 0; it < 8; ++it) {
      v8f acc = {};
#pragma unroll
      for (int ks = 0; ks < 4; ++ks) {
        v16bf aF = load_frag(&qbuf[p][it * 16 + lm][0], ks * 32, kh);
        acc = __builtin_amdgcn_wmma_f32_16x16x32_bf16(
            false, aF, false, bF[ks], (short)0, acc, false, false);
      }
      const float* mp = mrow + b * SS + ib * 128 + it * 16 + kh * 8;
      const float* cp = coef + b * SS + ib * 128 + it * 16 + kh * 8;
      float4 m0v = *(const float4*)(mp);
      float4 m1v = *(const float4*)(mp + 4);
      float4 c0v = *(const float4*)(cp);
      float4 c1v = *(const float4*)(cp + 4);
      float mv[8] = {m0v.x, m0v.y, m0v.z, m0v.w, m1v.x, m1v.y, m1v.z, m1v.w};
      float cv[8] = {c0v.x, c0v.y, c0v.z, c0v.w, c1v.x, c1v.y, c1v.z, c1v.w};
#pragma unroll
      for (int r = 0; r < 8; ++r)
        wloc += __expf(acc[r] * scale - mv[r]) * cv[r];  // partial col sum
    }
    __syncthreads();
  }
  wloc += __shfl_xor(wloc, 16, 32);  // combine the two halves (16 rows)
  if (kh == 0) wcol[b * SS + j0 + lm] = wloc;
}

// ---------------- K4: v[b,:] = sum_j w[b,j] * x[b,j,:] ----------------
__global__ __launch_bounds__(128) void out_reduce_kernel(
    const float* __restrict__ x, const float* __restrict__ wcol,
    float* __restrict__ out) {
  const int b = blockIdx.x, d = threadIdx.x;
  const float* xb = x + (size_t)b * SS * DD;
  const float* wb = wcol + b * SS;
  float acc = 0.0f;
  for (int j = 0; j < SS; ++j)
    acc += wb[j] * xb[(size_t)j * DD + d];
  out[b * DD + d] = acc;
}

extern "C" void kernel_launch(void* const* d_in, const int* in_sizes, int n_in,
                              void* d_out, int out_size, void* d_ws, size_t ws_size,
                              hipStream_t stream) {
  (void)in_sizes; (void)n_in; (void)out_size; (void)ws_size;
  const float* x     = (const float*)d_in[0];  // (B,S,D)
  const float* t     = (const float*)d_in[1];  // (B,S)
  const float* Wq    = (const float*)d_in[2];  // (D,U)
  const float* Wk    = (const float*)d_in[3];  // (D,U)
  const float* theta = (const float*)d_in[4];  // (1,S)
  const float* mu    = (const float*)d_in[5];  // (1,S)
  float* out = (float*)d_out;                  // (B,D)

  char* ws = (char*)d_ws;
  size_t off = 0;
  bf16_t* qk  = (bf16_t*)(ws + off); off += (size_t)2 * BB * SS * UU * sizeof(bf16_t); // 16 MB
  float* tf   = (float*)(ws + off);  off += (size_t)BB * SS * sizeof(float);
  float* mrow = (float*)(ws + off);  off += (size_t)BB * SS * sizeof(float);
  float* coef = (float*)(ws + off);  off += (size_t)BB * SS * sizeof(float);
  float* wcol = (float*)(ws + off);  off += (size_t)BB * SS * sizeof(float);

  timefact_kernel<<<dim3(BB * SS / 256), dim3(256), 0, stream>>>(t, theta, mu, tf);
  qk_proj_kernel<<<dim3(SS / 128, BB, 2), dim3(256), 0, stream>>>(x, Wq, Wk, qk);
  row_softmax_stats_kernel<<<dim3(SS / 128, BB), dim3(256), 0, stream>>>(qk, tf, mrow, coef);
  col_weight_kernel<<<dim3(SS / 128, BB), dim3(256), 0, stream>>>(qk, mrow, coef, wcol);
  out_reduce_kernel<<<dim3(BB), dim3(128), 0, stream>>>(x, wcol, out);
}
